// TripleViewNet_14886356647943
// MI455X (gfx1250) — compile-verified, matrix operand order checked by reading, not run on hardware
//
#include <hip/hip_runtime.h>
#include <hip/hip_bf16.h>
#include <math.h>

typedef _Float16 half_t;
typedef __attribute__((ext_vector_type(8)))  _Float16 v8h;
typedef __attribute__((ext_vector_type(16))) _Float16 v16h;
typedef __attribute__((ext_vector_type(8)))  float    v8f;

// ---------------------------------------------------------------- utilities
static __device__ __forceinline__ float atomicMaxF(float* addr, float val) {
  int* ai = (int*)addr;
  int old = __float_as_int(*addr);
  while (val > __int_as_float(old)) {
    int assumed = old;
    old = atomicCAS(ai, assumed, __float_as_int(val));
    if (old == assumed) break;
  }
  return __int_as_float(old);
}

__global__ void k_fill(float* __restrict__ p, float v, long n) {
  long i = (long)blockIdx.x * blockDim.x + threadIdx.x;
  if (i < n) p[i] = v;
}

__global__ void k_f32_to_f16(const float* __restrict__ src, half_t* __restrict__ dst, long n) {
  long i = (long)blockIdx.x * blockDim.x + threadIdx.x;
  if (i < n) dst[i] = (half_t)src[i];
}

// ---------------------------------------------------------------- WMMA GEMM
// C[N,64] = A[N,din] (f16) @ Bw[64,din]^T (f16, row-major = W), f32 accumulate.
// One wave computes a 16x64 tile: 4 accumulators, K-loop of 16x16x32 f16 WMMA.
__global__ void k_gemm_wmma(const half_t* __restrict__ A, const half_t* __restrict__ Bw,
                            float* __restrict__ C, int n_rows, int din) {
  int wave = threadIdx.x >> 5;
  int lane = threadIdx.x & 31;
  int tile = blockIdx.x * 8 + wave;
  int row0 = tile * 16;
  if (row0 >= n_rows) return;

  int m    = lane & 15;
  int kb   = (lane < 16) ? 0 : 8;      // wave32 f16 A/B fragment K-split
  int row  = row0 + m;
  bool rok = (row < n_rows);
  int ncol = lane & 15;

  v8f c0 = {}, c1 = {}, c2 = {}, c3 = {};

  for (int k0 = 0; k0 < din; k0 += 32) {
    v16h a;
    if (rok) {
      const half_t* ap = A + (long)row * din + k0 + kb;
      v8h alo = *(const v8h*)ap;          // K = kb .. kb+7
      v8h ahi = *(const v8h*)(ap + 16);   // K = 16+kb .. 16+kb+7
#pragma unroll
      for (int i = 0; i < 8; ++i) { a[i] = alo[i]; a[i + 8] = ahi[i]; }
    } else {
#pragma unroll
      for (int i = 0; i < 16; ++i) a[i] = (half_t)0.f;
    }
#pragma unroll
    for (int nt = 0; nt < 4; ++nt) {
      const half_t* bp = Bw + (long)(nt * 16 + ncol) * din + k0 + kb;
      v8h blo = *(const v8h*)bp;
      v8h bhi = *(const v8h*)(bp + 16);
      v16h bfrag;
#pragma unroll
      for (int i = 0; i < 8; ++i) { bfrag[i] = blo[i]; bfrag[i + 8] = bhi[i]; }
      v8f& cc = (nt == 0) ? c0 : (nt == 1) ? c1 : (nt == 2) ? c2 : c3;
      cc = __builtin_amdgcn_wmma_f32_16x16x32_f16(false, a, false, bfrag,
                                                  (short)0, cc, false, false);
    }
  }
#pragma unroll
  for (int r = 0; r < 8; ++r) {
    int mm = (lane < 16) ? r : (r + 8);
    int rr = row0 + mm;
    if (rr < n_rows) {
      float* cp = C + (long)rr * 64 + ncol;
      cp[0] = c0[r]; cp[16] = c1[r]; cp[32] = c2[r]; cp[48] = c3[r];
    }
  }
}

// ------------------------------------------------------------- GAT per-node
__global__ void k_attn(const float* __restrict__ hW, const float* __restrict__ a_src,
                       const float* __restrict__ a_dst, float* __restrict__ asrc,
                       float* __restrict__ adst, int n) {
  int i = blockIdx.x * blockDim.x + threadIdx.x;   // node*2 + head
  if (i >= n * 2) return;
  int node = i >> 1, h = i & 1;
  const float* hp = hW + (long)node * 64 + h * 32;
  const float* ws = a_src + h * 32;
  const float* wd = a_dst + h * 32;
  float sa = 0.f, sd = 0.f;
#pragma unroll
  for (int c = 0; c < 32; ++c) { float v = hp[c]; sa += v * ws[c]; sd += v * wd[c]; }
  asrc[i] = sa; adst[i] = sd;
}

static __device__ __forceinline__ void edge_sd(const int* __restrict__ ei, int E, int e,
                                               int& s, int& d) {
  if (e < E) { s = ei[e]; d = ei[E + e]; } else { s = d = e - E; }  // self loops
}

__global__ void k_edge_max(const int* __restrict__ ei, int E, int Etot,
                           const float* __restrict__ asrc, const float* __restrict__ adst,
                           float* __restrict__ mx) {
  int gid = blockIdx.x * blockDim.x + threadIdx.x;
  if (gid >= Etot * 2) return;
  int e = gid >> 1, h = gid & 1, s, d;
  edge_sd(ei, E, e, s, d);
  float ev = asrc[s * 2 + h] + adst[d * 2 + h];
  ev = ev > 0.f ? ev : 0.2f * ev;
  atomicMaxF(&mx[d * 2 + h], ev);
}

__global__ void k_edge_den(const int* __restrict__ ei, int E, int Etot,
                           const float* __restrict__ asrc, const float* __restrict__ adst,
                           const float* __restrict__ mx, float* __restrict__ den) {
  int gid = blockIdx.x * blockDim.x + threadIdx.x;
  if (gid >= Etot * 2) return;
  int e = gid >> 1, h = gid & 1, s, d;
  edge_sd(ei, E, e, s, d);
  float ev = asrc[s * 2 + h] + adst[d * 2 + h];
  ev = ev > 0.f ? ev : 0.2f * ev;
  atomicAdd(&den[d * 2 + h], __expf(ev - mx[d * 2 + h]));
}

__global__ void k_edge_agg(const int* __restrict__ ei, int E, int Etot,
                           const float* __restrict__ asrc, const float* __restrict__ adst,
                           const float* __restrict__ mx, const float* __restrict__ den,
                           const float* __restrict__ hW, float* __restrict__ agg) {
  int gid = blockIdx.x * blockDim.x + threadIdx.x;
  if (gid >= Etot * 2) return;
  int e = gid >> 1, h = gid & 1, s, d;
  edge_sd(ei, E, e, s, d);
  float ev = asrc[s * 2 + h] + adst[d * 2 + h];
  ev = ev > 0.f ? ev : 0.2f * ev;
  float ex = __expf(ev - mx[d * 2 + h]);
  float alpha = ex / (den[d * 2 + h] + 1e-16f);
  const float4* hs = (const float4*)(hW + (long)s * 64 + h * 32);
  float* op = agg + (long)d * 64 + h * 32;
#pragma unroll
  for (int i = 0; i < 8; ++i) {
    float4 v = hs[i];
    atomicAdd(op + i * 4 + 0, alpha * v.x);
    atomicAdd(op + i * 4 + 1, alpha * v.y);
    atomicAdd(op + i * 4 + 2, alpha * v.z);
    atomicAdd(op + i * 4 + 3, alpha * v.w);
  }
}

__global__ void k_bias_elu(float* __restrict__ agg, const float* __restrict__ bias, long n) {
  long i = (long)blockIdx.x * blockDim.x + threadIdx.x;
  if (i >= n) return;
  int c = (int)(i & 63);
  float v = agg[i] + bias[c];
  agg[i] = v > 0.f ? v : expm1f(v);
}

// ------------------------------------------------- LayerNorm(64) wave-per-node
__global__ void k_ln64(const float* __restrict__ in, float* __restrict__ out,
                       const float* __restrict__ g, const float* __restrict__ b, int n) {
  int wave = (blockIdx.x * blockDim.x + threadIdx.x) >> 5;
  int lane = threadIdx.x & 31;
  if (wave >= n) return;
  const float* row = in + (long)wave * 64;
  float v0 = row[lane], v1 = row[lane + 32];
  float s = v0 + v1;
#pragma unroll
  for (int o = 16; o > 0; o >>= 1) s += __shfl_xor(s, o, 32);
  float mu = s * (1.0f / 64.0f);
  float d0 = v0 - mu, d1 = v1 - mu;
  float q = d0 * d0 + d1 * d1;
#pragma unroll
  for (int o = 16; o > 0; o >>= 1) q += __shfl_xor(q, o, 32);
  float inv = rsqrtf(q * (1.0f / 64.0f) + 1e-5f);
  out[(long)wave * 64 + lane]      = d0 * inv * g[lane]      + b[lane];
  out[(long)wave * 64 + lane + 32] = d1 * inv * g[lane + 32] + b[lane + 32];
}

__global__ void k_gate(const float* __restrict__ hln, const float* __restrict__ gw,
                       const float* __restrict__ gb, float* __restrict__ gate, int n) {
  int i = blockIdx.x * blockDim.x + threadIdx.x;
  if (i >= n) return;
  const float* hp = hln + (long)i * 64;
  float s = gb[0];
#pragma unroll
  for (int c = 0; c < 64; ++c) s += hp[c] * gw[c];
  gate[i] = s;
}

__global__ void k_bmax(const float* __restrict__ gate, const int* __restrict__ batch,
                       float* __restrict__ mb, int n) {
  int i = blockIdx.x * blockDim.x + threadIdx.x;
  if (i < n) atomicMaxF(&mb[batch[i]], gate[i]);
}

__global__ void k_bexp(float* __restrict__ gate, const int* __restrict__ batch,
                       const float* __restrict__ mb, float* __restrict__ denb, int n) {
  int i = blockIdx.x * blockDim.x + threadIdx.x;
  if (i >= n) return;
  int b = batch[i];
  float ex = __expf(gate[i] - mb[b]);
  gate[i] = ex;
  atomicAdd(&denb[b], ex);
}

__global__ void k_bpool(const float* __restrict__ gate, const int* __restrict__ batch,
                        const float* __restrict__ denb, const float* __restrict__ hln,
                        float* __restrict__ pooled, int n) {
  int i = blockIdx.x * blockDim.x + threadIdx.x;
  if (i >= n) return;
  int b = batch[i];
  float sm = gate[i] / (denb[b] + 1e-16f);
  const float4* hp = (const float4*)(hln + (long)i * 64);
  float* op = pooled + (long)b * 64;
#pragma unroll
  for (int j = 0; j < 16; ++j) {
    float4 v = hp[j];
    atomicAdd(op + j * 4 + 0, sm * v.x);
    atomicAdd(op + j * 4 + 1, sm * v.y);
    atomicAdd(op + j * 4 + 2, sm * v.z);
    atomicAdd(op + j * 4 + 3, sm * v.w);
  }
}

// -------------------------------------------------- fused head (block/graph)
__global__ void k_fusion(const float* __restrict__ pooled, int Bn,
                         const float* __restrict__ ln_g, const float* __restrict__ ln_b,
                         const float* __restrict__ fw, const float* __restrict__ fb,
                         const float* __restrict__ ln2g, const float* __restrict__ ln2b,
                         const float* __restrict__ w1, const float* __restrict__ b1,
                         const float* __restrict__ w2, const float* __restrict__ b2,
                         float* __restrict__ out) {
  __shared__ float sc[192];
  __shared__ float sf[96];
  __shared__ float sz[64];
  __shared__ float stat[2];
  int b = blockIdx.x, t = threadIdx.x;     // 192 threads
  int view = t >> 6, c = t & 63;
  sc[t] = pooled[(long)view * Bn * 64 + (long)b * 64 + c];
  __syncthreads();
  if (t == 0) {
    float s = 0.f; for (int k = 0; k < 192; ++k) s += sc[k];
    float mu = s / 192.f, q = 0.f;
    for (int k = 0; k < 192; ++k) { float d = sc[k] - mu; q += d * d; }
    stat[0] = mu; stat[1] = rsqrtf(q / 192.f + 1e-5f);
  }
  __syncthreads();
  float cn = (sc[t] - stat[0]) * stat[1] * ln_g[t] + ln_b[t];
  __syncthreads();
  sc[t] = cn;
  __syncthreads();
  if (t < 96) {
    float s = fb[t];
    for (int k = 0; k < 192; ++k) s += sc[k] * fw[t * 192 + k];
    sf[t] = s;
  }
  __syncthreads();
  if (t == 0) {
    float s = 0.f; for (int k = 0; k < 96; ++k) s += sf[k];
    float mu = s / 96.f, q = 0.f;
    for (int k = 0; k < 96; ++k) { float d = sf[k] - mu; q += d * d; }
    stat[0] = mu; stat[1] = rsqrtf(q / 96.f + 1e-5f);
  }
  __syncthreads();
  if (t < 96) {
    float v = (sf[t] - stat[0]) * stat[1] * ln2g[t] + ln2b[t];
    sf[t] = fmaxf(v, 0.f);
  }
  __syncthreads();
  if (t < 64) {
    float s = b1[t];
    for (int k = 0; k < 96; ++k) s += sf[k] * w1[t * 96 + k];
    sz[t] = fmaxf(s, 0.f);
  }
  __syncthreads();
  if (t == 0) {
    float s = b2[0];
    for (int k = 0; k < 64; ++k) s += sz[k] * w2[k];
    out[b] = s;
  }
}

// ---------------------------------------------------------------- host side
static inline int cdiv(long a, int b) { return (int)((a + b - 1) / b); }

extern "C" void kernel_launch(void* const* d_in, const int* in_sizes, int n_in,
                              void* d_out, int out_size, void* d_ws, size_t ws_size,
                              hipStream_t stream) {
  (void)n_in; (void)ws_size;
  const int FD = 128, HD = 64, Bn = 256;
  const int N = in_sizes[0] / FD;
  const int E = in_sizes[1] / 2;
  const int Etot = E + N;                 // with self loops

  const float* x = (const float*)d_in[0];
  const int* ei[3] = { (const int*)d_in[1], (const int*)d_in[2], (const int*)d_in[3] };
  const int* batch = (const int*)d_in[4];
  auto P = [&](int i) { return (const float*)d_in[i]; };

  // ---- workspace carve (256B aligned) ----
  char* w = (char*)d_ws;
  auto carve = [&](size_t bytes) { void* p = (void*)w; w += (bytes + 255) & ~(size_t)255; return p; };
  half_t* h16   = (half_t*)carve((size_t)N * 128 * sizeof(half_t)); // also aliased as hln (f32, N*64)
  half_t* w16   = (half_t*)carve((size_t)64 * 128 * sizeof(half_t));
  float*  hW    = (float*) carve((size_t)N * 64 * sizeof(float));
  float*  agg   = (float*) carve((size_t)N * 64 * sizeof(float));
  float*  asrc  = (float*) carve((size_t)N * 2 * sizeof(float));
  float*  adst  = (float*) carve((size_t)N * 2 * sizeof(float));
  float*  mx    = (float*) carve((size_t)N * 2 * sizeof(float));
  float*  den   = (float*) carve((size_t)N * 2 * sizeof(float));
  float*  gate  = (float*) carve((size_t)N * sizeof(float));
  float*  pooled= (float*) carve((size_t)3 * Bn * 64 * sizeof(float));
  float*  mb    = (float*) carve((size_t)Bn * sizeof(float));
  float*  denb  = (float*) carve((size_t)Bn * sizeof(float));
  float*  hln   = (float*)h16;            // alias: free during post-GAT stage

  const int T = 256;
  k_fill<<<cdiv((long)3 * Bn * 64, T), T, 0, stream>>>(pooled, 0.f, (long)3 * Bn * 64);

  for (int v = 0; v < 3; ++v) {
    const int base = 5 + v * 16;
    const float* cur = x;
    int din = FD;
    for (int layer = 0; layer < 3; ++layer) {
      const int wi = base + layer * 4;
      // stage inputs to f16 (must read `cur` before zeroing agg)
      k_f32_to_f16<<<cdiv((long)N * din, T), T, 0, stream>>>(cur, h16, (long)N * din);
      k_f32_to_f16<<<cdiv(64L * din, T), T, 0, stream>>>(P(wi), w16, 64L * din);
      // h = x @ W^T via WMMA (16-row tiles, 8 waves/block)
      k_gemm_wmma<<<cdiv(cdiv(N, 16), 8), T, 0, stream>>>(h16, w16, hW, N, din);
      // per-node attention logit halves
      k_attn<<<cdiv((long)N * 2, T), T, 0, stream>>>(hW, P(wi + 1), P(wi + 2), asrc, adst, N);
      // scatter-softmax + aggregation (3 recompute passes, L2-resident)
      k_fill<<<cdiv((long)N * 2, T), T, 0, stream>>>(mx, -1e30f, (long)N * 2);
      k_fill<<<cdiv((long)N * 2, T), T, 0, stream>>>(den, 0.f, (long)N * 2);
      k_fill<<<cdiv((long)N * 64, T), T, 0, stream>>>(agg, 0.f, (long)N * 64);
      k_edge_max<<<cdiv((long)Etot * 2, T), T, 0, stream>>>(ei[v], E, Etot, asrc, adst, mx);
      k_edge_den<<<cdiv((long)Etot * 2, T), T, 0, stream>>>(ei[v], E, Etot, asrc, adst, mx, den);
      k_edge_agg<<<cdiv((long)Etot * 2, T), T, 0, stream>>>(ei[v], E, Etot, asrc, adst, mx, den, hW, agg);
      k_bias_elu<<<cdiv((long)N * 64, T), T, 0, stream>>>(agg, P(wi + 3), (long)N * 64);
      cur = agg;
      din = HD;
    }
    // LayerNorm + global attention pooling
    k_ln64<<<cdiv(N, 8), T, 0, stream>>>(agg, hln, P(base + 12), P(base + 13), N);
    k_gate<<<cdiv(N, T), T, 0, stream>>>(hln, P(base + 14), P(base + 15), gate, N);
    k_fill<<<cdiv(Bn, T), T, 0, stream>>>(mb, -1e30f, Bn);
    k_fill<<<cdiv(Bn, T), T, 0, stream>>>(denb, 0.f, Bn);
    k_bmax<<<cdiv(N, T), T, 0, stream>>>(gate, batch, mb, N);
    k_bexp<<<cdiv(N, T), T, 0, stream>>>(gate, batch, mb, denb, N);
    k_bpool<<<cdiv(N, T), T, 0, stream>>>(gate, batch, denb, hln, pooled + (long)v * Bn * 64, N);
  }

  // fusion + classifier head: one block per graph
  k_fusion<<<Bn, 192, 0, stream>>>(pooled, Bn,
                                   P(53), P(54), P(55), P(56), P(57), P(58),
                                   P(59), P(60), P(61), P(62), (float*)d_out);
}
